// Memory_51419348468246
// MI455X (gfx1250) — matrix operand order repaired and checked by reading, not compiled, not run on hardware
//
#include <hip/hip_runtime.h>
#include <stdint.h>

// ---------------------------------------------------------------------------
// Fused "memory network" kernel for MI455X (gfx1250, wave32, WMMA + TDM).
//   scores = q @ mempool^T   (bf16 WMMA 16x16x32, f32 accumulate)
//   att    = softmax(scores) (online, per-lane over half the items)
//   top-10 -> softmax(top10 att) -> out = sum_k w_k * mempool[idx_k]
// Mempool tiles are DMA'd into LDS with the Tensor Data Mover (TENSORcnt),
// overlapped with WMMA compute on the previous tile.
// ---------------------------------------------------------------------------

typedef __attribute__((ext_vector_type(16))) __bf16        v16bf;
typedef __attribute__((ext_vector_type(8)))  float         v8f;
typedef __attribute__((ext_vector_type(8)))  unsigned int  v8u;
typedef __attribute__((ext_vector_type(4)))  unsigned int  u32x4;
typedef __attribute__((ext_vector_type(8)))  int           i32x8;
typedef __attribute__((ext_vector_type(4)))  int           i32x4;

#define DIM       512
#define PSZ       1024              // H*W = 32*32
#define NUM_ITEM  2048
#define KTOP      10
#define NTILES    (NUM_ITEM / 16)   // 128 item tiles
#define QPW       16                // queries per wave
#define WAVES     8
#define QPB       (QPW * WAVES)     // 128 queries per block
#define APITCH    264               // dwords per item row in bf16 LDS tile
#define UPITCH    1088              // floats of scratch per wave (merge / out staging)
#define OPITCH    68                // floats per query row in out staging
#define PER_INPUT (16 * DIM * PSZ)  // 8388608 floats per input/output tensor
#define TILE_F32  (16 * DIM)        // 8192 floats = one contiguous 32KB mempool tile

#if __has_builtin(__builtin_amdgcn_tensor_load_to_lds) && \
    __has_builtin(__builtin_amdgcn_s_wait_tensorcnt)
#define HAVE_TDM 1
#else
#define HAVE_TDM 0
#endif

// pack two f32 -> packed bf16 pair (round-to-nearest-even)
static __device__ __forceinline__ unsigned int pack2bf(float a, float b) {
  unsigned int ua = __builtin_bit_cast(unsigned int, a);
  unsigned int ub = __builtin_bit_cast(unsigned int, b);
  ua += 0x7FFFu + ((ua >> 16) & 1u);
  ub += 0x7FFFu + ((ub >> 16) & 1u);
  return (ua >> 16) | (ub & 0xFFFF0000u);
}

#if HAVE_TDM
// Issue a TDM DMA: one contiguous 32KB mempool tile (8192 f32) -> LDS.
// D# built per CDNA5 ISA 8.3/8.4: 1-D tile, data_size=4B, type=2 ("image").
// This toolchain's builtin is the 6-arg form:
//   (uint32x4 g0, int32x8 g1, int32x4 g2, int32x4 g3, int32x8, i32 cpol)
static __device__ __forceinline__ void tdm_stage(const float* gsrc,
                                                 unsigned lds_off) {
  unsigned long long ga = (unsigned long long)(uintptr_t)gsrc;
  u32x4 g0 = {};
  g0[0] = 1u;                                            // count=1, user desc
  g0[1] = lds_off;                                       // lds_addr (bytes)
  g0[2] = (unsigned)(ga & 0xFFFFFFFFull);                // global_addr[31:0]
  g0[3] = (unsigned)((ga >> 32) & 0x1FFFFFFull)          // global_addr[56:32]
        | (2u << 30);                                    // type=2
  i32x8 g1 = {};
  g1[0] = (int)(2u << 16);        // workgroup_mask=0, data_size=2 (4 bytes)
  g1[1] = 0;                      // tensor_dim0[15:0]=0  (dim0 = 0x100000)
  g1[2] = (int)0x00010010u;       // tensor_dim0[31:16]=0x10, tensor_dim1=1
  g1[3] = (int)(8192u << 16);     // tile_dim0 = 8192 elements
  g1[4] = 0;                      // tile_dim1 = tile_dim2 = 0 (1-D tile)
  g1[5] = (int)0x00100000u;       // tensor_dim0_stride = 0x100000
  g1[6] = 0;
  g1[7] = 0;
  i32x4 gz = {};
  i32x8 gz8 = {};
  __builtin_amdgcn_tensor_load_to_lds(g0, g1, gz, gz, gz8, 0);
}
#endif

// LDS f32 tile -> LDS bf16 tile in A-fragment pair order
static __device__ __forceinline__ void convert_tile(const float* f32buf,
                                                    unsigned int* sA, int tid) {
  const int m = tid >> 4;             // item row 0..15
  const int ch0 = (tid & 15) * 32;    // 32 channels per thread
  const float4* s4 = (const float4*)(f32buf + m * DIM + ch0);
  float4 v[8];
#pragma unroll
  for (int i = 0; i < 8; ++i) v[i] = s4[i];
  unsigned int* dst = &sA[m * APITCH + (ch0 >> 1)];
#pragma unroll
  for (int i = 0; i < 8; ++i) {
    dst[2 * i]     = pack2bf(v[i].x, v[i].y);
    dst[2 * i + 1] = pack2bf(v[i].z, v[i].w);
  }
}

__global__ __launch_bounds__(256) void
memnet_fused_kernel(const float* __restrict__ x1,
                    const float* __restrict__ x2,
                    const float* __restrict__ mempool,
                    float* __restrict__ out)
{
  // bf16 mempool tile in A-fragment pair order: [16 items][APITCH dwords]
  __shared__ unsigned int sA[16 * APITCH];                       // 16896 B
  // phase-disjoint union: f32 DMA landing buffer (main loop, 32768 B)
  //                       merge/out staging    (epilogue, 34816 B)
  __shared__ __align__(16) unsigned char spool[WAVES * UPITCH * 4]; // 34816 B
  __shared__ float wqw[WAVES][16][KTOP];                         // 5120 B
  __shared__ int   wqi[WAVES][16][KTOP];                         // 5120 B
  float* f32buf = (float*)spool;                                 // total 61952 B

  const int tid  = threadIdx.x;
  const int lane = tid & 31;
  const int wid  = tid >> 5;

  // which input / which 128-query slab
  int gq = blockIdx.x * QPB;
  const float* xin;
  float* obase;
  if (gq < 16 * PSZ) { xin = x1; obase = out; }
  else               { xin = x2; obase = out + PER_INPUT; gq -= 16 * PSZ; }
  const int b  = gq >> 10;          // batch
  const int p0 = gq & (PSZ - 1);    // spatial base of this block
  const int pw = p0 + wid * QPW;    // spatial base of this wave

  // ------------------------------------------------------------------
  // Build B-fragments (queries, bf16) once; keep all K=512 in registers.
  // B layout (32x16, bf16): lane = query N (mod 16); lanes>=16 hold K+8;
// element e <-> K = 32*kc + (lane>=16 ? 8 : 0) + e + (e>=8 ? 8 : 0).
  // Channel stride in memory is PSZ, so lanes 0..15 coalesce over p.
  // ------------------------------------------------------------------
  v16bf qfrag[16];
  {
    const float* qx = xin + (size_t)b * DIM * PSZ + (size_t)(pw + (lane & 15));
    const int hb = (lane >= 16) ? 8 : 0;
#pragma unroll
    for (int kc = 0; kc < 16; ++kc) {
      v8u qa = {};
#pragma unroll
      for (int ep = 0; ep < 8; ++ep) {
        const int c0 = kc * 32 + hb + ep * 2 + (ep >= 4 ? 8 : 0);
        float f0 = qx[(size_t)c0 * PSZ];
        float f1 = qx[(size_t)(c0 + 1) * PSZ];
        qa[ep] = pack2bf(f0, f1);
      }
      qfrag[kc] = __builtin_bit_cast(v16bf, qa);
    }
  }

  // per-lane state: this lane covers items {j + 8*(lane>=16)} of each tile
  // for query (lane & 15): online softmax (rm, rs) + sorted top-10.
  float rm = -1e30f, rs = 0.0f;
  float tv[KTOP];
  int   ti[KTOP];
#pragma unroll
  for (int k = 0; k < KTOP; ++k) { tv[k] = -1e30f; ti[k] = 0; }

  const int abase = (lane & 15) * APITCH + ((lane >> 4) & 1) * 4;
  const int hi8   = (lane >> 4) << 3;
  const unsigned ldsoff = (unsigned)(uintptr_t)(void*)f32buf;

  // ---- staging helpers (TDM path: wave 0 issues the DMA; EXEC ignored) ----
#if HAVE_TDM
#define STAGE(tile)  do { if (wid == 0) tdm_stage(mempool + (size_t)(tile) * TILE_F32, ldsoff); } while (0)
#define STAGE_WAIT() do { if (wid == 0) __builtin_amdgcn_s_wait_tensorcnt(0); } while (0)
#else
  // fallback: cooperative synchronous copy (256 threads x 8 float4)
#define STAGE(tile)  do {                                                     \
    const float4* s4 = (const float4*)(mempool + (size_t)(tile) * TILE_F32);  \
    float4* d4 = (float4*)f32buf;                                             \
    _Pragma("unroll")                                                         \
    for (int i = 0; i < 8; ++i) d4[tid + i * 256] = s4[tid + i * 256];        \
  } while (0)
#define STAGE_WAIT() do { } while (0)
#endif

  // prologue: tile 0 -> LDS, convert, then start DMA of tile 1
  STAGE(0);
  STAGE_WAIT();
  __syncthreads();
  convert_tile(f32buf, sA, tid);
  __syncthreads();
  STAGE(1);

  for (int t = 0; t < NTILES; ++t) {
    // ---- 16 x v_wmma_f32_16x16x32_bf16 : D[item][query] over K=512 ----
    // A-fragment LDS loads are double-buffered so kc+1 loads issue
    // before the kc wait (LDS latency hides under the WMMA).
    v8f acc = {};
    {
      const unsigned int* sAb = &sA[abase];
      uint4 c0 = *(const uint4*)(sAb + 0);
      uint4 c1 = *(const uint4*)(sAb + 8);
#pragma unroll
      for (int kc = 0; kc < 16; ++kc) {
        uint4 n0 = c0, n1 = c1;
        if (kc < 15) {
          n0 = *(const uint4*)(sAb + (kc + 1) * 16);
          n1 = *(const uint4*)(sAb + (kc + 1) * 16 + 8);
        }
        v8u au = {c0.x, c0.y, c0.z, c0.w, c1.x, c1.y, c1.z, c1.w};
        v16bf a = __builtin_bit_cast(v16bf, au);
        acc = __builtin_amdgcn_wmma_f32_16x16x32_bf16(
            false, a, false, qfrag[kc], (short)0, acc, false, false);
        c0 = n0; c1 = n1;
      }
    }

    // ---- online softmax over this tile's 8 scores ----
    float tm = acc[0];
#pragma unroll
    for (int j = 1; j < 8; ++j) tm = fmaxf(tm, acc[j]);
    float nm  = fmaxf(rm, tm);
    float add = 0.0f;
#pragma unroll
    for (int j = 0; j < 8; ++j) add += __expf(acc[j] - nm);
    rs = rs * __expf(rm - nm) + add;
    rm = nm;

    // ---- top-10 insert (common case: 1 compare) ----
#pragma unroll
    for (int j = 0; j < 8; ++j) {
      float v = acc[j];
      if (v > tv[KTOP - 1]) {
        tv[KTOP - 1] = v;
        ti[KTOP - 1] = t * 16 + j + hi8;
#pragma unroll
        for (int r = KTOP - 1; r >= 1; --r) {
          if (tv[r] > tv[r - 1]) {
            float tf = tv[r]; tv[r] = tv[r - 1]; tv[r - 1] = tf;
            int   tn = ti[r]; ti[r] = ti[r - 1]; ti[r - 1] = tn;
          }
        }
      }
    }

    __syncthreads();                    // everyone done reading sA
    if (t + 1 < NTILES) {
      STAGE_WAIT();                     // f32buf now holds tile t+1
      __syncthreads();                  // publish to all waves
      convert_tile(f32buf, sA, tid);    // f32 -> bf16 fragment order
      __syncthreads();                  // sA ready; f32buf free
      if (t + 2 < NTILES) STAGE(t + 2); // DMA overlaps compute of t+1
    }
  }

  // ------------------------------------------------------------------
  // Merge the two lane halves per query, double softmax, final weights.
  // (spool is re-used as merge/out scratch from here on.)
  // ------------------------------------------------------------------
  float* uwork = (float*)spool;
  float* wbase = uwork + wid * UPITCH;
  {
    float* mp = wbase + lane * 24;
#pragma unroll
    for (int k = 0; k < KTOP; ++k) {
      mp[k]        = tv[k];
      mp[KTOP + k] = __int_as_float(ti[k]);
    }
    mp[20] = rm;
    mp[21] = rs;
  }
  __syncthreads();
  {
    const int q = lane & 15;
    const float* A = wbase + q * 24;
    const float* B = wbase + (q + 16) * 24;
    float mv[KTOP]; int mi[KTOP];
    int ia = 0, ib = 0;
#pragma unroll
    for (int k = 0; k < KTOP; ++k) {            // merge two sorted 10-lists
      float va = A[ia], vb = B[ib];
      if (va >= vb) { mv[k] = va; mi[k] = __float_as_int(A[KTOP + ia]); ++ia; }
      else          { mv[k] = vb; mi[k] = __float_as_int(B[KTOP + ib]); ++ib; }
    }
    float ma = A[20], sa = A[21], mb = B[20], sb = B[21];
    float M    = fmaxf(ma, mb);
    float S    = sa * __expf(ma - M) + sb * __expf(mb - M);
    float invS = 1.0f / S;
    float att0 = __expf(mv[0] - M) * invS;      // largest att (list sorted desc)
    float wv[KTOP];
    float esum = 0.0f;
#pragma unroll
    for (int k = 0; k < KTOP; ++k) {            // softmax over top-10 att values
      float att = __expf(mv[k] - M) * invS;
      wv[k] = __expf(att - att0);
      esum += wv[k];
    }
    float inv = 1.0f / esum;
    if (lane < 16) {
#pragma unroll
      for (int k = 0; k < KTOP; ++k) {
        wqw[wid][q][k] = wv[k] * inv;
        wqi[wid][q][k] = mi[k];
      }
    }
  }
  __syncthreads();

  // ------------------------------------------------------------------
  // out[n, :] = sum_k w_k * mempool[idx_k, :]
  // Staged through LDS in 64-channel chunks so that both the mempool row
  // reads (coalesced 128B, L2-resident) and the channel-strided output
  // stores (lane over p -> 64B segments) are coalesced.
  // ------------------------------------------------------------------
  float* outp = obase + (size_t)b * DIM * PSZ + pw;
  for (int cb = 0; cb < 8; ++cb) {
    const int ch0 = cb * 64 + lane * 2;
#pragma unroll 1
    for (int q = 0; q < 16; ++q) {
      float a0 = 0.0f, a1 = 0.0f;
#pragma unroll
      for (int k = 0; k < KTOP; ++k) {
        float w = wqw[wid][q][k];
        const float* mr = mempool + (size_t)wqi[wid][q][k] * DIM + ch0;
        a0 += w * mr[0];
        a1 += w * mr[1];
      }
      wbase[q * OPITCH + lane * 2]     = a0;
      wbase[q * OPITCH + lane * 2 + 1] = a1;
    }
    __syncthreads();
#pragma unroll
    for (int it = 0; it < 32; ++it) {
      const int cl = it * 2 + (lane >> 4);
      const int p  = lane & 15;
      outp[(size_t)(cb * 64 + cl) * PSZ + p] = wbase[p * OPITCH + cl];
    }
    __syncthreads();
  }
}

extern "C" void kernel_launch(void* const* d_in, const int* in_sizes, int n_in,
                              void* d_out, int out_size, void* d_ws, size_t ws_size,
                              hipStream_t stream) {
  (void)in_sizes; (void)n_in; (void)out_size; (void)d_ws; (void)ws_size;
  const float* x1 = (const float*)d_in[0];
  const float* x2 = (const float*)d_in[1];
  const float* mp = (const float*)d_in[2];
  float* out = (float*)d_out;
  // 2 * 16384 queries / 128 queries-per-block = 256 blocks of 256 threads
  memnet_fused_kernel<<<dim3(256), dim3(256), 0, stream>>>(x1, x2, mp, out);
}
